// DxKnowledgeEncoder_44444321579591
// MI455X (gfx1250) — matrix operand order repaired and checked by reading, not compiled, not run on hardware
//
#include <hip/hip_runtime.h>
#include <hip/hip_bf16.h>
#include <hip/hip_fp16.h>

typedef __attribute__((ext_vector_type(16))) _Float16 v16h;
typedef __attribute__((ext_vector_type(8)))  _Float16 v8h;
typedef __attribute__((ext_vector_type(8)))  float    v8f;
typedef unsigned int u32x4 __attribute__((ext_vector_type(4)));
typedef int          i32x4 __attribute__((ext_vector_type(4)));
typedef int          i32x8 __attribute__((ext_vector_type(8)));

#define VERY_NEG (-1e30f)
#define SHUF16(lo,hi) __builtin_shufflevector((lo),(hi),0,1,2,3,4,5,6,7,8,9,10,11,12,13,14,15)
#define WMMA_F16(a,b,c) __builtin_amdgcn_wmma_f32_16x16x32_f16(false,(a),false,(b),(short)0,(c),false,false)

#if defined(__has_builtin)
#if __has_builtin(__builtin_amdgcn_tensor_load_to_lds) && __has_builtin(__builtin_amdgcn_s_wait_tensorcnt)
#define HAVE_TDM 1
#endif
#endif

__device__ __forceinline__ v8f vzero8() {
    v8f z;
#pragma unroll
    for (int i = 0; i < 8; ++i) z[i] = 0.f;
    return z;
}

// A-matrix fragment (16x32 f16, MxK): lane<16 -> K 0..15,
// lane>=16 -> K 16..31, row = lane&15. Two contiguous 16B loads.
__device__ __forceinline__ v16h frag_a(const _Float16* base, int stride, int rowBase, int lane) {
    int r  = rowBase + (lane & 15);
    int kb = (lane >> 4) * 16;
    const _Float16* p = base + r * stride + kb;
    v8h lo = *(const v8h*)(p);
    v8h hi = *(const v8h*)(p + 8);
    return SHUF16(lo, hi);
}

// B-matrix fragment (32x16 f16, KxN) from a pre-swizzled 512-half tile
// [n16][k32]: lane<16 -> K 0..15, lane>=16 -> K 16..31, col = lane&15.
// One contiguous 32B read per lane (works for global or LDS pointers).
__device__ __forceinline__ v16h frag_b_tile(const _Float16* tileBase, int lane) {
    const _Float16* p = tileBase + (lane & 15) * 32 + (lane >> 4) * 16;
    v8h lo = *(const v8h*)(p);
    v8h hi = *(const v8h*)(p + 8);
    return SHUF16(lo, hi);
}

// Same B layout but from a stride-40 LDS tile staged as [n][k] (attn K / V^T).
__device__ __forceinline__ v16h frag_b_s40(const _Float16* base, int rowBase, int lane) {
    int r  = rowBase + (lane & 15);
    int kb = (lane >> 4) * 16;
    const _Float16* p = base + r * 40 + kb;
    v8h lo = *(const v8h*)(p);
    v8h hi = *(const v8h*)(p + 8);
    return SHUF16(lo, hi);
}

// ---------------------------------------------------------------- utilities
__global__ void f32_to_f16_kernel(const float* __restrict__ s, _Float16* __restrict__ d, int n) {
    for (int i = blockIdx.x * blockDim.x + threadIdx.x; i < n; i += gridDim.x * blockDim.x)
        d[i] = (_Float16)s[i];
}

// Pack weight W[K][N] (f32) into WMMA B-fragment tiles:
// out[((k/32)*(N/16) + n/16)*512 + (n&15)*32 + (k&31)] (f16)
__global__ void pack_b_kernel(const float* __restrict__ w, _Float16* __restrict__ out,
                              int K, int N) {
    int total = K * N;
    for (int i = blockIdx.x * blockDim.x + threadIdx.x; i < total; i += gridDim.x * blockDim.x) {
        int k = i / N, n = i - k * N;
        size_t o = ((size_t)(k >> 5) * (N >> 4) + (n >> 4)) * 512 + (n & 15) * 32 + (k & 31);
        out[o] = (_Float16)w[i];
    }
}

__global__ void zero_row_kernel(float* __restrict__ p) {
    p[threadIdx.x] = 0.f;
}

__global__ void gather_kernel(const int* __restrict__ ids, const float* __restrict__ embIn,
                              const float* __restrict__ dict, float* __restrict__ xin,
                              float* __restrict__ xdag, int total) {
    for (int i = blockIdx.x * blockDim.x + threadIdx.x; i < total; i += gridDim.x * blockDim.x) {
        int row = i >> 8, hcol = i & 255;
        int id = ids[row];
        xin[i]  = embIn[(size_t)id * 256 + hcol];
        xdag[i] = dict[(size_t)id * 256 + hcol];
    }
}

// ---------------------------------------------------------- generic WMMA GEMM
// out[M,N] = op( A[M,K](f32) @ B(f16, pre-swizzled tiles) + bias + residual ).
// 64x64 block tile, 128 threads (4 waves); wave w owns M-subtile w, N-subtiles 0..3.
// A staging is wave-private; B fragments stream straight from L2-resident
// swizzled weights: no block barriers needed.
__global__ __launch_bounds__(128)
void gemm_f16w(const float* __restrict__ A, const _Float16* __restrict__ Bsw,
               const float* __restrict__ bias, const float* __restrict__ resid,
               float* __restrict__ out, int M, int N, int K, int relu) {
    __shared__ __attribute__((aligned(16))) _Float16 As[64 * 40];
    const int bm = blockIdx.y * 64, bn = blockIdx.x * 64;
    const int tid = threadIdx.x, wave = tid >> 5, lane = tid & 31;
    const int ntilesN = N >> 4;

    v8f acc[4];
#pragma unroll
    for (int i = 0; i < 4; ++i) acc[i] = vzero8();

    for (int k0 = 0; k0 < K; k0 += 32) {
        {   // wave-private A stage: rows 16*wave..16*wave+15, f32 -> f16
            int r = tid >> 1, cb = (tid & 1) * 16;
            const float* s = A + (size_t)(bm + r) * K + k0 + cb;
            _Float16* d = As + r * 40 + cb;
#pragma unroll
            for (int j = 0; j < 16; ++j) d[j] = (_Float16)s[j];
            if (k0 + 32 < K) __builtin_prefetch((const void*)(s + 32), 0, 1);
        }
        v16h a = frag_a(As, 40, wave * 16, lane);
        const _Float16* Bkt = Bsw + ((size_t)(k0 >> 5) * ntilesN + (bn >> 4)) * 512;
#pragma unroll
        for (int nt = 0; nt < 4; ++nt) {
            v16h b = frag_b_tile(Bkt + nt * 512, lane);
            acc[nt] = WMMA_F16(a, b, acc[nt]);
        }
    }

    const int hi8 = (lane >> 4) * 8, col = lane & 15;
#pragma unroll
    for (int nt = 0; nt < 4; ++nt) {
        int n = bn + nt * 16 + col;
        float bv = bias ? bias[n] : 0.f;
#pragma unroll
        for (int r = 0; r < 8; ++r) {
            int m = bm + wave * 16 + hi8 + r;
            float v = acc[nt][r] + bv;
            if (resid) v += resid[(size_t)m * N + n];
            if (relu) v = fmaxf(v, 0.f);
            out[(size_t)m * N + n] = v;
        }
    }
}

// ------------------------------------------------------------- DAG embedding
// One block = 8 codes (64 rows of A=8 ancestors). GEMM1 (K=512,N=256) via WMMA;
// the per-K-step 16KB swizzled w1 chunk (shared by all 8 waves) is moved into
// LDS by the Tensor Data Mover; fused bias+relu, score, softmax, weighted sum.
__global__ __launch_bounds__(256)
void dag_kernel(const int* __restrict__ leaves, const int* __restrict__ anc,
                const float* __restrict__ msk,
                const _Float16* __restrict__ embH, const float* __restrict__ embF,
                const _Float16* __restrict__ w1sw, const float* __restrict__ b1,
                const float* __restrict__ w2, const float* __restrict__ b2,
                float* __restrict__ dict) {
    __shared__ int   idxL[64], idxA[64];
    __shared__ float mk[64];
    __shared__ __attribute__((aligned(16))) _Float16 As[64 * 40];
    __shared__ __attribute__((aligned(16))) _Float16 Bs[16 * 512];   // 16 N-tiles
    __shared__ __attribute__((aligned(16))) _Float16 xs[64 * 256];
    __shared__ float sPart[64][4];
    __shared__ float sArr[64];
    __shared__ float wgt[64];

    const int tid = threadIdx.x, wave = tid >> 5, lane = tid & 31;
    const int c0 = blockIdx.x * 8;

    if (tid < 64) {
        int c = c0 + (tid >> 3), a = tid & 7;
        idxL[tid] = leaves[c * 8 + a];
        idxA[tid] = anc[c * 8 + a];
        mk[tid]   = msk[c * 8 + a];
    }
    __syncthreads();

    v8f acc[4][2];
#pragma unroll
    for (int i = 0; i < 4; ++i)
#pragma unroll
        for (int j = 0; j < 2; ++j) acc[i][j] = vzero8();

    for (int k0 = 0; k0 < 512; k0 += 32) {
        {   // stage gathered (leaves||ancestors)*mask rows as f16
            int row = tid >> 2, cb = (tid & 3) * 8;
            float m = mk[row];
            int il = idxL[row], ia = idxA[row];
            _Float16* d = As + row * 40 + cb;
#pragma unroll
            for (int j = 0; j < 8; ++j) {
                int kg = k0 + cb + j;
                _Float16 e = (kg < 256) ? embH[(size_t)il * 256 + kg]
                                        : embH[(size_t)ia * 256 + (kg - 256)];
                d[j] = (m != 0.f) ? e : (_Float16)0.f;
            }
        }
#if defined(HAVE_TDM)
        if (wave == 0) {   // TDM: DMA this K-step's 16KB swizzled w1 chunk to LDS
            unsigned long long ga =
                (unsigned long long)(uintptr_t)(w1sw + (size_t)(k0 >> 5) * 8192);
            unsigned int ldsOff = (unsigned int)(uintptr_t)(&Bs[0]);
            u32x4 g0;
            g0[0] = 1u;                                  // count=1 valid descriptor
            g0[1] = ldsOff;                              // lds_addr (bytes)
            g0[2] = (unsigned int)ga;                    // global_addr[31:0]
            g0[3] = (unsigned int)((ga >> 32) & 0x01FFFFFFu) | (2u << 30); // type=2
            i32x8 g1;
            g1[0] = (int)(1u << 16);                     // data_size=1 (2B/elem)
            g1[1] = (int)(8192u << 16);                  // tensor_dim0[15:0]=8192
            g1[2] = (int)(1u << 16);                     // tensor_dim1=1
            g1[3] = (int)(8192u << 16);                  // tile_dim0=8192
            g1[4] = 1;                                   // tile_dim1=1
            g1[5] = 8192;                                // tensor_dim0_stride
            g1[6] = 0;
            g1[7] = 0;
            i32x4 z4;
            z4[0] = 0; z4[1] = 0; z4[2] = 0; z4[3] = 0;
#if __clang_major__ >= 23
            i32x8 z8;
#pragma unroll
            for (int q = 0; q < 8; ++q) z8[q] = 0;
            __builtin_amdgcn_tensor_load_to_lds(g0, g1, z4, z4, z8, 0);
#else
            __builtin_amdgcn_tensor_load_to_lds(g0, g1, z4, z4, 0);
#endif
            __builtin_amdgcn_s_wait_tensorcnt(0);
        }
#else
        {   // fallback: cooperative vector copy of the 16KB chunk
            const v8h* s = (const v8h*)(w1sw + (size_t)(k0 >> 5) * 8192);
            v8h* d = (v8h*)Bs;
#pragma unroll
            for (int j = 0; j < 4; ++j) d[tid * 4 + j] = s[tid * 4 + j];
        }
#endif
        __syncthreads();
#pragma unroll
        for (int mt = 0; mt < 4; ++mt) {
            v16h a = frag_a(As, 40, mt * 16, lane);
#pragma unroll
            for (int j = 0; j < 2; ++j) {
                v16h b = frag_b_tile(Bs + (wave * 2 + j) * 512, lane);
                acc[mt][j] = WMMA_F16(a, b, acc[mt][j]);
            }
        }
        __syncthreads();
    }

    {   // x = relu(acc + b1) -> f16 LDS
        const int hi8 = (lane >> 4) * 8, col = lane & 15;
#pragma unroll
        for (int mt = 0; mt < 4; ++mt)
#pragma unroll
            for (int j = 0; j < 2; ++j) {
                int n = (wave * 2 + j) * 16 + col;
                float bv = b1[n];
#pragma unroll
                for (int r = 0; r < 8; ++r) {
                    int m = mt * 16 + hi8 + r;
                    float v = acc[mt][j][r] + bv;
                    xs[m * 256 + n] = (_Float16)fmaxf(v, 0.f);
                }
            }
    }
    __syncthreads();

    {   // s = x @ w2 (partial per 64-col segment)
        int row = tid >> 2, seg = tid & 3;
        float p = 0.f;
        for (int i = 0; i < 64; ++i) {
            int n = seg * 64 + i;
            p += (float)xs[row * 256 + n] * w2[n];
        }
        sPart[row][seg] = p;
    }
    __syncthreads();
    if (tid < 64)
        sArr[tid] = sPart[tid][0] + sPart[tid][1] + sPart[tid][2] + sPart[tid][3]
                  + b2[0] + (1.f - mk[tid]) * VERY_NEG;
    __syncthreads();
    if (tid < 8) {   // per-code softmax over 8 ancestors
        float mx = -3.4e38f;
        for (int a = 0; a < 8; ++a) mx = fmaxf(mx, sArr[tid * 8 + a]);
        float e[8], sm = 0.f;
        for (int a = 0; a < 8; ++a) { e[a] = __expf(sArr[tid * 8 + a] - mx); sm += e[a]; }
        for (int a = 0; a < 8; ++a) wgt[tid * 8 + a] = (e[a] / sm) * mk[tid * 8 + a];
    }
    __syncthreads();

    {   // dag_emb[c] = sum_a wgt * ancestors_f32, write to dict row c+1
        int cl = tid >> 5, hb = (tid & 31) * 8;
        float o[8];
#pragma unroll
        for (int j = 0; j < 8; ++j) o[j] = 0.f;
        for (int a = 0; a < 8; ++a) {
            float w = wgt[cl * 8 + a];
            if (w != 0.f) {
                const float* s = embF + (size_t)idxA[cl * 8 + a] * 256 + hb;
#pragma unroll
                for (int j = 0; j < 8; ++j) o[j] += w * s[j];
            }
        }
        float* d = dict + (size_t)(c0 + cl + 1) * 256 + hb;
#pragma unroll
        for (int j = 0; j < 8; ++j) d[j] = o[j];
    }
}

// ----------------------------------------------------- multi-head attention
__global__ __launch_bounds__(128)
void attn_kernel(const float* __restrict__ Q, const float* __restrict__ Kg,
                 const float* __restrict__ V, const float* __restrict__ cm,
                 float* __restrict__ CTX) {
    __shared__ __attribute__((aligned(16))) _Float16 qs[32 * 40];
    __shared__ __attribute__((aligned(16))) _Float16 ks[32 * 40];
    __shared__ __attribute__((aligned(16))) _Float16 vt[32 * 40];
    __shared__ float sc[32 * 256];
    __shared__ __attribute__((aligned(16))) _Float16 pl[32 * 256];
    __shared__ float red[32][4];
    __shared__ float rowM[32], rowS[32];

    const int tid = threadIdx.x, wave = tid >> 5, lane = tid & 31;
    const int bid = blockIdx.x;
    const int qt = bid & 7, h = (bid >> 3) & 7, b = bid >> 6;
    const size_t base = ((size_t)b * 256) * 256 + h * 32;
    const int mt = wave >> 1, nt = wave & 1;
    const float scale = 0.17677669529663689f;  // 1/sqrt(32)

    {   // stage Q tile [32 q x 32 d] f16
        int r = tid >> 2, db = (tid & 3) * 8;
        const float* s = Q + base + (size_t)(qt * 32 + r) * 256 + db;
#pragma unroll
        for (int j = 0; j < 8; ++j) qs[r * 40 + db + j] = (_Float16)s[j];
    }

    for (int kt = 0; kt < 8; ++kt) {
        {   // stage K tile [32 key x 32 d]; serves directly as B^T fragments
            int r = tid >> 2, db = (tid & 3) * 8;
            const float* s = Kg + base + (size_t)(kt * 32 + r) * 256 + db;
#pragma unroll
            for (int j = 0; j < 8; ++j) ks[r * 40 + db + j] = (_Float16)s[j];
        }
        __syncthreads();
        v16h a = frag_a(qs, 40, mt * 16, lane);
        v16h bb = frag_b_s40(ks, nt * 16, lane);
        v8f c = vzero8();
        c = WMMA_F16(a, bb, c);
        {   // scaled + masked scores -> LDS
            int hi8 = (lane >> 4) * 8, col = lane & 15;
            int key = kt * 32 + nt * 16 + col;
            float mterm = (1.f - cm[b * 256 + key]) * VERY_NEG;
#pragma unroll
            for (int r = 0; r < 8; ++r)
                sc[(mt * 16 + hi8 + r) * 256 + key] = c[r] * scale + mterm;
        }
        __syncthreads();
    }

    {   // row max
        int row = tid >> 2, seg = tid & 3;
        float mx = -3.4e38f;
        for (int i = 0; i < 64; ++i) mx = fmaxf(mx, sc[row * 256 + seg * 64 + i]);
        red[row][seg] = mx;
    }
    __syncthreads();
    if (tid < 32) rowM[tid] = fmaxf(fmaxf(red[tid][0], red[tid][1]),
                                    fmaxf(red[tid][2], red[tid][3]));
    __syncthreads();
    {   // exp + partial sums (in place)
        int row = tid >> 2, seg = tid & 3;
        float m = rowM[row], sm = 0.f;
        for (int i = 0; i < 64; ++i) {
            float e = __expf(sc[row * 256 + seg * 64 + i] - m);
            sc[row * 256 + seg * 64 + i] = e;
            sm += e;
        }
        red[row][seg] = sm;
    }
    __syncthreads();
    if (tid < 32) rowS[tid] = red[tid][0] + red[tid][1] + red[tid][2] + red[tid][3];
    __syncthreads();
    {   // normalized probs -> f16
        int row = tid >> 2, seg = tid & 3;
        float inv = 1.f / rowS[row];
        for (int i = 0; i < 64; ++i)
            pl[row * 256 + seg * 64 + i] = (_Float16)(sc[row * 256 + seg * 64 + i] * inv);
    }
    __syncthreads();

    // ctx = P @ V, K-loop over 256 keys
    v8f acc = vzero8();
    for (int kk = 0; kk < 8; ++kk) {
        {   // stage V chunk transposed: vt[d][key]
            int dd = tid >> 2, kb = (tid & 3) * 8;
#pragma unroll
            for (int j = 0; j < 8; ++j)
                vt[dd * 40 + kb + j] =
                    (_Float16)V[base + (size_t)(kk * 32 + kb + j) * 256 + dd];
        }
        __syncthreads();
        v16h a = frag_a(pl + kk * 32, 256, mt * 16, lane);
        v16h bb = frag_b_s40(vt, nt * 16, lane);
        acc = WMMA_F16(a, bb, acc);
        __syncthreads();
    }
    {
        int hi8 = (lane >> 4) * 8, col = lane & 15;
#pragma unroll
        for (int r = 0; r < 8; ++r)
            CTX[base + (size_t)(qt * 32 + mt * 16 + hi8 + r) * 256 + nt * 16 + col] = acc[r];
    }
}

// ---------------------------------------------------------- attention pooling
__global__ __launch_bounds__(256)
void pool_kernel(const float* __restrict__ X, const float* __restrict__ cm,
                 const float* __restrict__ pw, const float* __restrict__ pb,
                 float* __restrict__ out) {
    __shared__ float aw[256];
    __shared__ float red2[2];
    const int b = blockIdx.x, t = threadIdx.x;
    const float* xb = X + (size_t)b * 65536;
    float s = 0.f;
    for (int h = 0; h < 256; ++h) s += xb[t * 256 + h] * pw[h];
    s += pb[0] + (1.f - cm[b * 256 + t]) * VERY_NEG;
    aw[t] = s;
    __syncthreads();
    if (t == 0) {
        float mx = -3.4e38f;
        for (int i = 0; i < 256; ++i) mx = fmaxf(mx, aw[i]);
        red2[0] = mx;
    }
    __syncthreads();
    float e = __expf(s - red2[0]);
    aw[t] = e;
    __syncthreads();
    if (t == 0) {
        float sm = 0.f;
        for (int i = 0; i < 256; ++i) sm += aw[i];
        red2[1] = sm;
    }
    __syncthreads();
    float a = e / red2[1];
    aw[t] = a;
    __syncthreads();
    float acc = 0.f;
    for (int s2 = 0; s2 < 256; ++s2) acc += aw[s2] * xb[s2 * 256 + t];
    out[b * 256 + t] = acc;
}

// -------------------------------------------------------------------- driver
extern "C" void kernel_launch(void* const* d_in, const int* in_sizes, int n_in,
                              void* d_out, int out_size, void* d_ws, size_t ws_size,
                              hipStream_t stream) {
    (void)in_sizes; (void)n_in; (void)out_size; (void)ws_size;
    const int B = 128, S = 256, H = 256, C = 20000, NODES = 25000;
    const int BS = B * S;                      // 32768
    const size_t XBYTES = (size_t)BS * H * 4;  // 33,554,432

    const int*   input_ids  = (const int*)d_in[0];
    const float* code_mask  = (const float*)d_in[1];
    const int*   leaves     = (const int*)d_in[2];
    const int*   anc        = (const int*)d_in[3];
    const float* dmask      = (const float*)d_in[4];
    const float* embInit    = (const float*)d_in[5];
    const float* embInputs  = (const float*)d_in[6];
    const float* attn_w1    = (const float*)d_in[7];
    const float* attn_b1    = (const float*)d_in[8];
    const float* attn_w2    = (const float*)d_in[9];
    const float* attn_b2    = (const float*)d_in[10];
    const float* pool_w     = (const float*)d_in[11];
    const float* pool_b     = (const float*)d_in[12];

    size_t off = 0;
    char* wsb = (char*)d_ws;
    auto alloc = [&](size_t bytes) -> char* {
        char* p = wsb + off;
        off += (bytes + 255) & ~(size_t)255;
        return p;
    };
    const int embN = (NODES + 1) * H;
    _Float16* embH = (_Float16*)alloc((size_t)embN * 2);
    _Float16* w1H  = (_Float16*)alloc((size_t)512 * 256 * 2);
    _Float16* encW[2][6];
    for (int e = 0; e < 2; ++e) {
        for (int j = 0; j < 4; ++j) encW[e][j] = (_Float16*)alloc((size_t)H * H * 2);
        encW[e][4] = (_Float16*)alloc((size_t)H * 4 * H * 2);
        encW[e][5] = (_Float16*)alloc((size_t)4 * H * H * 2);
    }
    float* dict = (float*)alloc((size_t)(C + 1) * H * 4);
    float* XIN0 = (float*)alloc(XBYTES);
    float* XIN1 = (float*)alloc(XBYTES);
    float* Qb   = (float*)alloc(XBYTES);   // Q,K,V,CTX contiguous; FFH aliases them
    float* Kb   = (float*)alloc(XBYTES);
    float* Vb   = (float*)alloc(XBYTES);
    float* CTXb = (float*)alloc(XBYTES);
    float* FFH  = Qb;
    float* X1   = (float*)alloc(XBYTES);
    float* DAGH = (float*)alloc(XBYTES);

    auto conv = [&](const float* s, _Float16* d, int n) {
        int blocks = (n + 255) / 256;
        if (blocks > 4096) blocks = 4096;
        f32_to_f16_kernel<<<blocks, 256, 0, stream>>>(s, d, n);
    };
    auto pack = [&](const float* s, _Float16* d, int Kd, int Nd) {
        int blocks = (Kd * Nd + 255) / 256;
        if (blocks > 4096) blocks = 4096;
        pack_b_kernel<<<blocks, 256, 0, stream>>>(s, d, Kd, Nd);
    };

    conv(embInit, embH, embN);
    pack(attn_w1, w1H, 512, 256);
    for (int e = 0; e < 2; ++e) {
        const int wb = 13 + e * 8;
        for (int j = 0; j < 4; ++j) pack((const float*)d_in[wb + j], encW[e][j], H, H);
        pack((const float*)d_in[wb + 4], encW[e][4], H, 4 * H);
        pack((const float*)d_in[wb + 6], encW[e][5], 4 * H, H);
    }

    zero_row_kernel<<<1, 256, 0, stream>>>(dict);
    dag_kernel<<<C / 8, 256, 0, stream>>>(leaves, anc, dmask, embH, embInit,
                                          w1H, attn_b1, attn_w2, attn_b2, dict);

    gather_kernel<<<4096, 256, 0, stream>>>(input_ids, embInputs, dict, XIN0, XIN1, BS * H);

    float* out = (float*)d_out;
    dim3 g256(H / 64, BS / 64);
    dim3 g1024(4 * H / 64, BS / 64);
    for (int e = 0; e < 2; ++e) {
        const float* x   = (e == 0) ? XIN0 : XIN1;
        const float* fb1 = (const float*)d_in[13 + e * 8 + 5];
        const float* fb2 = (const float*)d_in[13 + e * 8 + 7];
        float* outPtr    = (e == 0) ? out : DAGH;

        gemm_f16w<<<g256, 128, 0, stream>>>(x, encW[e][0], nullptr, nullptr, Qb, BS, H, H, 0);
        gemm_f16w<<<g256, 128, 0, stream>>>(x, encW[e][1], nullptr, nullptr, Kb, BS, H, H, 0);
        gemm_f16w<<<g256, 128, 0, stream>>>(x, encW[e][2], nullptr, nullptr, Vb, BS, H, H, 0);
        attn_kernel<<<B * 8 * 8, 128, 0, stream>>>(Qb, Kb, Vb, code_mask, CTXb);
        gemm_f16w<<<g256, 128, 0, stream>>>(CTXb, encW[e][3], nullptr, x, X1, BS, H, H, 0);
        gemm_f16w<<<g1024, 128, 0, stream>>>(X1, encW[e][4], fb1, nullptr, FFH, BS, 4 * H, H, 1);
        gemm_f16w<<<g256, 128, 0, stream>>>(FFH, encW[e][5], fb2, X1, outPtr, BS, H, 4 * H, 0);
    }

    pool_kernel<<<B, 256, 0, stream>>>(DAGH, code_mask, pool_w, pool_b, out + (size_t)BS * H);
}
